// LearnableTensorProduct_65944927862932
// MI455X (gfx1250) — matrix operand order repaired and checked by reading, not compiled, not run on hardware
//
#include <hip/hip_runtime.h>
#include <hip/hip_bf16.h>
#include <cmath>
#include <cstring>
#include <cstdint>
#include <complex>

typedef __attribute__((ext_vector_type(16))) _Float16 v16h;
typedef __attribute__((ext_vector_type(8)))  float    v8f;

#define ROW      288      // 32 channels * 9 irreps floats
#define ROW4     72       // ROW / 4
#define OUT_ROW  944      // 64*1 + 64*3 + 64*5 + 32*7 + 16*9
#define MAXE     512
#define NFRAG    63       // total WMMA A-fragments across all l
#define LDS_BYTES (2*16*ROW*4 + 15360*2)   // x1/x2 tile (36864B) + f16 feat (30720B)

struct Ranges { int s[5]; int e[5]; };
struct Ent    { uint32_t m; float v; };    // packed CG entry: one b64 load

// A-fragment element -> K offset within a 32-wide chunk (16-bit A 16x32 layout):
// lanes 0-15 cover K {0-7,16-23}, lanes 16-31 the same +8.
__device__ __host__ inline int afrag_bk(int e) {
    int r = e >> 1, h = e & 1;
    return (r < 4) ? (2*r + h) : (16 + 2*(r-4) + h);
}

// ---------------------------------------------------------------------------
// Prep kernel: W (f32, row-major) -> scaled f16 in WMMA A-fragment order.
// One block per fragment; frag layout in ws: [frag][lane][e] halves.
// ---------------------------------------------------------------------------
__global__ __launch_bounds__(256)
void prep_w_kernel(const float* __restrict__ W0, const float* __restrict__ W1,
                   const float* __restrict__ W2, const float* __restrict__ W3,
                   const float* __restrict__ W4, _Float16* __restrict__ wf)
{
    const int frag = blockIdx.x;   // 0..62
    const float* W; int K, base, nk; float scl;
    if (frag < 12)      { W = W0; K = 96;  base = 0;  nk = 3; scl = 0.1020620726f; }
    else if (frag < 32) { W = W1; K = 160; base = 12; nk = 5; scl = 0.0790569415f; }
    else if (frag < 56) { W = W2; K = 192; base = 32; nk = 6; scl = 0.0721687837f; }
    else if (frag < 62) { W = W3; K = 96;  base = 56; nk = 3; scl = 0.1020620726f; }
    else                { W = W4; K = 32;  base = 62; nk = 1; scl = 0.1767766953f; }
    const int f  = frag - base;
    const int ot = f / nk;
    const int kc = f - ot * nk;
    for (int idx = threadIdx.x; idx < 512; idx += 256) {
        const int lane = idx >> 4, e = idx & 15;
        const int kk   = kc*32 + e + ((lane >= 16) ? 16 : 0);
        const int arow = ot*16 + (lane & 15);
        wf[(size_t)frag*512 + idx] = (_Float16)(W[arow*K + kk] * scl);
    }
}

// ---------------------------------------------------------------------------
// Fused kernel: per 16-sample tile, for each output l:
//   phase A: sparse CG tensor product -> feat (f16) in LDS
//   phase B: per-(otile, component) GEMM via v_wmma_f32_16x16x32_f16,
//            A = one v16h load from pre-swizzled ws, B = one v16h load from LDS
// ---------------------------------------------------------------------------
__global__ __launch_bounds__(256)
void tp_wmma_fused_kernel(const float* __restrict__ x1, const float* __restrict__ x2,
                          const _Float16* __restrict__ wf,
                          float* __restrict__ out,
                          const Ent* __restrict__ ents,
                          Ranges rg, int N)
{
    extern __shared__ __align__(16) char smem[];
    float*    x1s  = (float*)smem;               // [16][288]
    float*    x2s  = x1s + 16*ROW;               // [16][288]
    _Float16* feat = (_Float16*)(x2s + 16*ROW);  // [comp][chanHi][sample16][chanLo16] f16

    const int n0 = blockIdx.x * 16;

    // ---- stage the 16-sample input tile into LDS (b128 loads/stores) ----
    {
        const float4* x1v = (const float4*)x1;
        const float4* x2v = (const float4*)x2;
        float4* s1v = (float4*)x1s;
        float4* s2v = (float4*)x2s;
        const float4 z = make_float4(0.f, 0.f, 0.f, 0.f);
        for (int t = threadIdx.x; t < 16*ROW4; t += 256) {
            int r = t / ROW4, c = t - r*ROW4;
            int row = n0 + r;
            bool ok = row < N;
            s1v[t] = ok ? x1v[(size_t)row*ROW4 + c] : z;
            s2v[t] = ok ? x2v[(size_t)row*ROW4 + c] : z;
        }
    }
    // prefetch next tile (gfx1250 global_prefetch_b8; speculative, OOB dropped)
    if (n0 + 16 < N) {
        const char* p = (const char*)(x1 + (size_t)(n0+16)*ROW);
        __builtin_prefetch(p + (threadIdx.x << 6), 0, 1);
    }
    __syncthreads();

    const int lane = threadIdx.x & 31;
    const int wave = threadIdx.x >> 5;
    const int col  = lane & 15;   // sample within tile (B/D column), and A row M
    const int kh   = lane >> 4;   // lane-half selector

    #pragma unroll 1
    for (int l = 0; l < 5; ++l) {
        int K, mt, obase, cb, fbase, nk;
        switch (l) {
            case 0: K=96;  mt=4; obase=0;   cb=6;  fbase=0;  nk=3; break;
            case 1: K=160; mt=4; obase=64;  cb=10; fbase=12; nk=5; break;
            case 2: K=192; mt=4; obase=256; cb=12; fbase=32; nk=6; break;
            case 3: K=96;  mt=2; obase=576; cb=6;  fbase=56; nk=3; break;
            default:K=32;  mt=1; obase=800; cb=2;  fbase=62; nk=1; break;
        }
        const int comps = 2*l + 1;

        // ---- phase A: tensor product for this l -> feat in LDS ----
        // 512 (sample,u) jobs over 256 threads; entries grouped (path,k), last flagged
        for (int jj = threadIdx.x; jj < 512; jj += 256) {
            int s = jj >> 5, u = jj & 31;
            const float* a1 = x1s + s*ROW + u*9;
            const float* a2 = x2s + s*ROW + u*9;
            float acc = 0.f;
            const int e1 = rg.e[l];
            for (int e = rg.s[l]; e < e1; ++e) {
                Ent q = ents[e];
                acc = fmaf(q.v, a1[q.m & 15u] * a2[(q.m >> 4) & 15u], acc);
                if (q.m & 256u) {
                    // dest = base(path,k) + chanHi(u)*256 + sample*16 + chanLo(u)
                    feat[(q.m >> 9) + ((u >> 4) << 8) + (s << 4) + (u & 15)] = (_Float16)acc;
                    acc = 0.f;
                }
            }
        }
        __syncthreads();

        // ---- phase B: D[o,sample] += Wf16 x feat, f16 WMMA / f32 acc ----
        int job = 0;
        for (int ot = 0; ot < mt; ++ot) {
            for (int c = 0; c < comps; ++c) {
                if ((job++ & 7) != wave) continue;
                v8f acc = {};
                for (int k0 = 0; k0 < K; k0 += 32) {
                    // A fragment: pre-swizzled, one 32B-aligned v16h load
                    const _Float16* ap = wf + ((size_t)(fbase + ot*nk + (k0 >> 5)) << 9)
                                            + (lane << 4);
                    v16h a = *(const v16h*)ap;
                    // B fragment: lanes 0-15 K=k0..k0+15, lanes 16-31 K=k0+16..k0+31
                    const _Float16* bp = feat + ((((c*cb + (k0 >> 4) + kh) * 16) + col) << 4);
                    v16h b = *(const v16h*)bp;
                    acc = __builtin_amdgcn_wmma_f32_16x16x32_f16(
                              false, a, false, b, (short)0, acc, false, false);
                }
                // D: VGPR r -> M=r (lanes 0-15) / M=r+8 (lanes 16-31), N=col
                const int srow = n0 + col;
                if (srow < N) {
                    float* po = out + (size_t)srow*OUT_ROW + obase
                              + (ot*16 + (kh ? 8 : 0))*comps + c;
                    #pragma unroll
                    for (int r = 0; r < 8; ++r) po[r*comps] = acc[r];
                }
            }
        }
        __syncthreads();   // feat reused by next l
    }
}

// ---------------------------------------------------------------------------
// Host: exact mirror of the reference Wigner-3j construction
// ---------------------------------------------------------------------------
namespace cgbuild {
typedef std::complex<double> cd;

static double fct(int n) { double r = 1.0; for (int i = 2; i <= n; ++i) r *= i; return r; }

static double su2cg(int j1, int j2, int j3, int m1, int m2, int m3) {
    if (m3 != m1 + m2 || j3 < std::abs(j1 - j2) || j3 > j1 + j2) return 0.0;
    double pref = std::sqrt((2.0*j3 + 1.0) * fct(j3+j1-j2) * fct(j3-j1+j2) * fct(j1+j2-j3)
                            / fct(j1+j2+j3+1));
    pref *= std::sqrt(fct(j3+m3)*fct(j3-m3)*fct(j1-m1)*fct(j1+m1)*fct(j2-m2)*fct(j2+m2));
    double s = 0.0;
    for (int k = 0; k <= j1 + j2 - j3; ++k) {
        int a[6] = {k, j1+j2-j3-k, j1-m1-k, j2+m2-k, j3-j2+m1+k, j3-j1-m2+k};
        bool bad = false; for (int t = 0; t < 6; ++t) if (a[t] < 0) bad = true;
        if (bad) continue;
        double d = 1.0; for (int t = 0; t < 6; ++t) d *= fct(a[t]);
        s += ((k & 1) ? -1.0 : 1.0) / d;
    }
    return pref * s;
}

static void qmat(int l, cd q[9][9]) {
    for (int i = 0; i < 9; ++i) for (int j = 0; j < 9; ++j) q[i][j] = 0.0;
    const double r = 1.0 / std::sqrt(2.0);
    for (int m = -l; m < 0; ++m) { q[l+m][l-m] = r; q[l+m][l+m] = cd(0.0, -r); }
    q[l][l] = 1.0;
    for (int m = 1; m <= l; ++m) {
        double sg = (m & 1) ? -1.0 : 1.0;
        q[l+m][l+m] = sg * r; q[l+m][l-m] = cd(0.0, sg * r);
    }
    cd ph(1.0, 0.0); for (int t = 0; t < l; ++t) ph *= cd(0.0, -1.0);
    for (int i = 0; i < 2*l+1; ++i) for (int j = 0; j < 2*l+1; ++j) q[i][j] *= ph;
}

static void w3j(int l1, int l2, int l3, double T[9][9][9]) {
    int n1 = 2*l1+1, n2 = 2*l2+1, n3 = 2*l3+1;
    double C[9][9][9]; std::memset(C, 0, sizeof(C));
    for (int m1 = -l1; m1 <= l1; ++m1)
        for (int m2 = -l2; m2 <= l2; ++m2) {
            int m3 = m1 + m2;
            if (m3 < -l3 || m3 > l3) continue;
            C[l1+m1][l2+m2][l3+m3] = su2cg(l1, l2, l3, m1, m2, m3);
        }
    cd q1[9][9], q2[9][9], q3[9][9];
    qmat(l1, q1); qmat(l2, q2); qmat(l3, q3);
    static cd Tc[9][9][9];
    double sre = 0.0, sim = 0.0;
    for (int a = 0; a < n1; ++a)
        for (int b = 0; b < n2; ++b)
            for (int c = 0; c < n3; ++c) {
                cd t(0.0, 0.0);
                for (int i = 0; i < n1; ++i)
                    for (int j = 0; j < n2; ++j)
                        for (int k = 0; k < n3; ++k)
                            if (C[i][j][k] != 0.0)
                                t += q1[a][i] * q2[b][j] * std::conj(q3[c][k]) * C[i][j][k];
                Tc[a][b][c] = t;
                sre += std::fabs(t.real()); sim += std::fabs(t.imag());
            }
    double inv = 1.0 / std::sqrt((double)n3);
    std::memset(T, 0, sizeof(double)*9*9*9);
    for (int a = 0; a < n1; ++a)
        for (int b = 0; b < n2; ++b)
            for (int c = 0; c < n3; ++c)
                T[a][b][c] = ((sre >= sim) ? Tc[a][b][c].real() : Tc[a][b][c].imag()) * inv;
}

// Sparse entries sorted by output l3; groups are (path,k)-contiguous, last flagged (bit 8)
static int build(Ent* ents, Ranges* rg) {
    const int off[3] = {0, 1, 4};
    const int cb[5]  = {6, 10, 12, 6, 2};   // K/16 per l
    int n = 0;
    for (int l3 = 0; l3 < 5; ++l3) {
        rg->s[l3] = n;
        int p = 0;
        for (int l1 = 0; l1 < 3; ++l1)
            for (int l2 = 0; l2 < 3; ++l2) {
                if (l3 < std::abs(l1 - l2) || l3 > l1 + l2) continue;
                static double T[9][9][9];
                w3j(l1, l2, l3, T);
                for (int k = 0; k < 2*l3 + 1; ++k) {
                    int destBase = k * cb[l3] * 256 + p * 512;   // + chanHi(u)*256 + s*16 + chanLo(u)
                    int start = n;
                    for (int i = 0; i < 2*l1 + 1; ++i)
                        for (int j = 0; j < 2*l2 + 1; ++j) {
                            double cv = T[i][j][k];
                            if (std::fabs(cv) > 1e-9 && n < MAXE) {
                                ents[n].m = (uint32_t)(off[l1]+i) | ((uint32_t)(off[l2]+j) << 4)
                                          | ((uint32_t)destBase << 9);
                                ents[n].v = (float)cv;
                                ++n;
                            }
                        }
                    if (n == start) {              // keep every slot written
                        if (n < MAXE) { ents[n].m = ((uint32_t)destBase << 9) | 256u; ents[n].v = 0.f; ++n; }
                    } else {
                        ents[n-1].m |= 256u;
                    }
                }
                ++p;
            }
        rg->e[l3] = n;
    }
    return n;
}
} // namespace cgbuild

extern "C" void kernel_launch(void* const* d_in, const int* in_sizes, int n_in,
                              void* d_out, int out_size, void* d_ws, size_t ws_size,
                              hipStream_t stream)
{
    (void)n_in; (void)out_size; (void)ws_size;

    // Static so the host buffer stays valid across graph replays of the HtoD copy.
    // Rebuilt deterministically (same values) on every call.
    static Ent h_ents[MAXE];
    Ranges rg;
    cgbuild::build(h_ents, &rg);

    const float* x1 = (const float*)d_in[0];
    const float* x2 = (const float*)d_in[1];
    const float* W0 = (const float*)d_in[2];
    const float* W1 = (const float*)d_in[3];
    const float* W2 = (const float*)d_in[4];
    const float* W3 = (const float*)d_in[5];
    const float* W4 = (const float*)d_in[6];
    float* out = (float*)d_out;
    const int N = in_sizes[0] / ROW;

    // ws layout: [0,4096): Ent table; [4096, 4096 + 63*512*2): f16 A-fragments of W
    hipMemcpyAsync(d_ws, h_ents, sizeof(h_ents), hipMemcpyHostToDevice, stream);

    const Ent*  dents = (const Ent*)d_ws;
    _Float16*   dwf   = (_Float16*)((char*)d_ws + sizeof(h_ents));

    prep_w_kernel<<<NFRAG, 256, 0, stream>>>(W0, W1, W2, W3, W4, dwf);

    const int blocks = (N + 15) / 16;
    tp_wmma_fused_kernel<<<blocks, 256, LDS_BYTES, stream>>>(
        x1, x2, dwf, out, dents, rg, N);
}
